// HMR_76690936037394
// MI455X (gfx1250) — matrix operand order repaired and checked by reading, not compiled, not run on hardware
//
#include <hip/hip_runtime.h>
#include <hip/hip_bf16.h>
#include <stdint.h>

typedef __bf16 bf16t;
typedef __attribute__((ext_vector_type(16))) __bf16 v16bf;
typedef __attribute__((ext_vector_type(8)))  float    v8f;
typedef __attribute__((ext_vector_type(4)))  unsigned int u32x4;
typedef __attribute__((ext_vector_type(8)))  int i32x8;
typedef __attribute__((ext_vector_type(4)))  int i32x4;

#define H1     1024
#define BB     16
#define TT     128
#define BT     (BB*TT)        // 2048
#define NGATE  7
#define NBIG   (NGATE*H1)     // 7168
#define KBIG   (3*H1)         // 3072
#define NSMALL (NBIG + 2*H1)  // 9216
#define KSMALL (2*H1)         // 2048
#define REC    6

#if __has_builtin(__builtin_amdgcn_tensor_load_to_lds)
#define TDM_PATH 1
#else
#define TDM_PATH 0
#endif

__device__ __forceinline__ float sigf(float x){ return 1.0f/(1.0f+__expf(-x)); }

#if TDM_PATH
// Issue one TDM 2D tile load: 128 rows x 32 bf16 (contiguous rows of 64B into LDS).
// D# group0: count=1 | lds_addr | 57b global_addr | type=2
// D# group1: data_size=2B, tensor_dim0/1, tile_dim0=32, tile_dim1=128, 48b dim0 stride
__device__ __forceinline__ void tdm_load_tile(unsigned lds_byte, const void* gptr,
                                              unsigned tensor_w, unsigned tensor_h,
                                              unsigned long long stride_elems){
  unsigned long long ga = (unsigned long long)(uintptr_t)gptr;
  u32x4 g0 = { 1u,                                  // count=1 (valid), user mode
               lds_byte,                            // lds_addr
               (unsigned)(ga & 0xFFFFFFFFu),        // global_addr[31:0]
               (unsigned)((ga >> 32) & 0x1FFFFFFu)  // global_addr[56:32]
                 | 0x80000000u };                   // type=2 ("image")
  i32x8 g1 = {
    (int)(1u << 16),                                             // wg_mask=0, data_size=1 (2B)
    (int)((tensor_w & 0xFFFFu) << 16),                           // abar=0 | tensor_dim0[15:0]
    (int)(((tensor_w >> 16) & 0xFFFFu) | ((tensor_h & 0xFFFFu) << 16)),
    (int)(((tensor_h >> 16) & 0xFFFFu) | (32u << 16)),           // tile_dim0 = 32
    (int)(128u),                                                 // tile_dim1=128, tile_dim2=0
    (int)(unsigned)(stride_elems & 0xFFFFFFFFu),                 // dim0_stride[31:0]
    (int)(unsigned)((stride_elems >> 32) & 0xFFFFu),             // dim0_stride[47:32]
    0
  };
  i32x4 z4 = {0, 0, 0, 0};
#if defined(__clang_major__) && (__clang_major__ >= 23)
  i32x8 z8 = {0, 0, 0, 0, 0, 0, 0, 0};
  __builtin_amdgcn_tensor_load_to_lds(g0, g1, z4, z4, z8, 0);
#else
  __builtin_amdgcn_tensor_load_to_lds(g0, g1, z4, z4, 0);
#endif
}
#endif

// ---------------- one-time prep ----------------

// Wb[n][ks]: n = g*1024+kcol (7 gates: f,l,r,q,i,o,g_f), ks = seg*1024+kk
__global__ void k_prep_wb(const float* __restrict__ U, const float* __restrict__ Wlr,
                          const float* __restrict__ W, const float* __restrict__ gWp,
                          bf16t* __restrict__ Wb){
  size_t i = (size_t)blockIdx.x*blockDim.x + threadIdx.x;
  if (i >= (size_t)NBIG*KBIG) return;
  int ks = (int)(i % KBIG);
  int n  = (int)(i / KBIG);
  int g = n >> 10, kcol = n & 1023;
  int seg = ks >> 10, kk = ks & 1023;
  float w = 0.f;
  if (g < 6) {
    if (seg == 0) {
      int gw = (g == 5) ? 4 : g;   // source bug: o-gate uses Wi
      w = U[((size_t)g<<20) + ((size_t)kk<<10) + kcol]
        + W[((size_t)gw<<20) + ((size_t)kk<<10) + kcol];
    } else if (seg == 1) {
      w = Wlr[((size_t)g<<21) + ((size_t)kk<<10) + kcol];
    } else {
      w = Wlr[((size_t)g<<21) + ((size_t)(kk+1024)<<10) + kcol];
    }
  } else {
    if (seg == 0) w = gWp[((size_t)kk<<10) + kcol];
  }
  Wb[i] = (bf16t)w;
}

__global__ void k_prep_wsmall(const float* __restrict__ Z, const float* __restrict__ gZ,
                              const float* __restrict__ gWp, bf16t* __restrict__ Wsm){
  size_t i = (size_t)blockIdx.x*blockDim.x + threadIdx.x;
  if (i >= (size_t)NSMALL*KSMALL) return;
  int ks = (int)(i % KSMALL);
  int n  = (int)(i / KSMALL);
  int seg = ks >> 10, kk = ks & 1023;
  float w = 0.f;
  if (n < NBIG){
    if (seg == 0){
      int g = n >> 10, kcol = n & 1023;
      w = (g < 6) ? Z[((size_t)g<<20) + ((size_t)kk<<10) + kcol]
                  : gZ[((size_t)kk<<10) + kcol];
    }
  } else if (n < NBIG + H1){
    int kcol = n - NBIG;
    w = (seg==0) ? gZ[((size_t)1<<20) + ((size_t)kk<<10) + kcol]
                 : gWp[((size_t)1<<20) + ((size_t)kk<<10) + kcol];
  } else {
    int kcol = n - NBIG - H1;
    w = (seg==0) ? gZ[((size_t)2<<20) + ((size_t)kk<<10) + kcol]
                 : gWp[((size_t)2<<20) + ((size_t)kk<<10) + kcol];
  }
  Wsm[i] = (bf16t)w;
}

__global__ void k_prep_bsmall(const float* __restrict__ b6, const float* __restrict__ gb,
                              float* __restrict__ bsmall){
  int n = blockIdx.x*blockDim.x + threadIdx.x;
  if (n >= NSMALL) return;
  float v;
  if      (n < 6*H1)      v = b6[n];
  else if (n < NBIG)      v = gb[n - 6*H1];
  else if (n < NBIG+H1)   v = gb[H1 + (n - NBIG)];
  else                    v = gb[2*H1 + (n - NBIG - H1)];
  bsmall[n] = v;
}

__global__ void k_copy2(const float* __restrict__ a, const float* __restrict__ b,
                        float* __restrict__ oa, float* __restrict__ ob, int n){
  int i = blockIdx.x*blockDim.x + threadIdx.x;
  if (i < n){ oa[i] = a[i]; ob[i] = b[i]; }
}

__global__ void k_means2(const float* __restrict__ h, const float* __restrict__ c,
                         float* __restrict__ g, float* __restrict__ cg){
  int i = blockIdx.x*blockDim.x + threadIdx.x;
  if (i >= BB*H1) return;
  int b = i >> 10, k = i & 1023;
  const float* ph = h + ((size_t)b*TT)*H1 + k;
  const float* pc = c + ((size_t)b*TT)*H1 + k;
  float s0 = 0.f, s1 = 0.f;
  for (int t = 0; t < TT; ++t){ s0 += ph[(size_t)t*H1]; s1 += pc[(size_t)t*H1]; }
  g[i]  = s0 * (1.f/TT);
  cg[i] = s1 * (1.f/TT);
}

// ---------------- per-step ----------------

__global__ void k_winsum(const float* __restrict__ h, const float* __restrict__ c,
                         bf16t* __restrict__ hbf, bf16t* __restrict__ hbbf,
                         bf16t* __restrict__ habf,
                         float* __restrict__ chb, float* __restrict__ cha){
  size_t i = (size_t)blockIdx.x*blockDim.x + threadIdx.x;
  if (i >= (size_t)BT*H1) return;
  int t = (int)((i >> 10) & (TT-1));
  float hv = h[i];
  float hb = 0.f, ha = 0.f, cb = 0.f, ca = 0.f;
  #pragma unroll
  for (int s = 1; s <= 2; ++s){
    if (t - s >= 0){ hb += h[i - (size_t)s*H1]; cb += c[i - (size_t)s*H1]; }
    if (t + s < TT){ ha += h[i + (size_t)s*H1]; ca += c[i + (size_t)s*H1]; }
  }
  hbf[i]  = (bf16t)hv;
  hbbf[i] = (bf16t)hb;
  habf[i] = (bf16t)ha;
  chb[i] = cb; cha[i] = ca;
}

__global__ void k_build_gA(const float* __restrict__ g, const float* __restrict__ h,
                           bf16t* __restrict__ gA){
  int i = blockIdx.x*blockDim.x + threadIdx.x;
  if (i >= BB*H1) return;
  int b = i >> 10, k = i & 1023;
  const float* ph = h + ((size_t)b*TT)*H1 + k;
  float s = 0.f;
  for (int t = 0; t < TT; ++t) s += ph[(size_t)t*H1];
  gA[(size_t)b*KSMALL + k]      = (bf16t)g[i];
  gA[(size_t)b*KSMALL + H1 + k] = (bf16t)(s * (1.f/TT));
}

// small GEMM: [16 x 2048] x [2048 x 9216] -> smallout, + bias (one wave per N16 tile)
__global__ __launch_bounds__(256) void k_gemm_small(const bf16t* __restrict__ gA,
                                                    const bf16t* __restrict__ Wsm,
                                                    const float* __restrict__ bsmall,
                                                    float* __restrict__ out){
  int wave = threadIdx.x >> 5, lane = threadIdx.x & 31;
  int nBase = (blockIdx.x*8 + wave) * 16;
  int m = lane & 15;
  int koffA = (lane < 16) ? 0 : 8;
  int koffB = (lane < 16) ? 0 : 8;
  v8f acc = {};
  const bf16t* pa = gA  + (size_t)m*KSMALL;
  const bf16t* pb = Wsm + (size_t)(nBase + m)*KSMALL;
  union Frag { v16bf v; u32x4 q[2]; };
  for (int kc = 0; kc < KSMALL/32; ++kc){
    Frag af, bfv;
    af.q[0]  = *(const u32x4*)(pa + koffA);
    af.q[1]  = *(const u32x4*)(pa + koffA + 16);
    bfv.q[0] = *(const u32x4*)(pb + koffB);
    bfv.q[1] = *(const u32x4*)(pb + koffB + 16);
    acc = __builtin_amdgcn_wmma_f32_16x16x32_bf16(false, af.v, false, bfv.v,
                                                  (short)0, acc, false, false);
    pa += 32; pb += 32;
  }
  int col   = nBase + (lane & 15);
  int rbase = (lane >> 4) * 8;
  float bias = bsmall[col];
  #pragma unroll
  for (int r = 0; r < 8; ++r){
    int row = rbase + r;
    out[(size_t)row*NSMALL + col] = acc[r] + bias;
  }
}

// main GEMM: [2048 x 3072(h|hb|ha)] x [3072 x 7168] -> pre, + zterm broadcast.
// 128x128 block tile, 8 waves (2x4), wave = 64x32 = 8 wmma tiles.
// Staging: TDM tensor_load_to_lds (wave 0 issues, TENSORcnt), double buffered.
__global__ __launch_bounds__(256) void k_gemm_main(const bf16t* __restrict__ Ah,
                                                   const bf16t* __restrict__ Abm,
                                                   const bf16t* __restrict__ Aam,
                                                   const bf16t* __restrict__ Wb,
                                                   const float* __restrict__ zterm,
                                                   float* __restrict__ pre){
  __shared__ __align__(16) bf16t As[2][128][32];
  __shared__ __align__(16) bf16t Bs[2][128][32];
  const int tid  = threadIdx.x;
  const int wave = tid >> 5, lane = tid & 31;
  const int wm = wave >> 2, wn = wave & 3;
  const int mBase = blockIdx.x * 128;          // == batch tile (T==128)
  const int nBase = blockIdx.y * 128;

  const bf16t* Aseg[3] = {Ah, Abm, Aam};
  const int NK = KBIG/32;                      // 96 chunks

  auto stage = [&](int kc, int buf){
    int ksBase = kc * 32;
    int seg = ksBase >> 10, kk = ksBase & 1023;
#if TDM_PATH
    if (wave == 0){
      tdm_load_tile((unsigned)(uintptr_t)(&As[buf][0][0]),
                    Aseg[seg] + (size_t)mBase*H1 + kk, H1, BT, H1);
      tdm_load_tile((unsigned)(uintptr_t)(&Bs[buf][0][0]),
                    Wb + (size_t)nBase*KBIG + ksBase, KBIG, NBIG, KBIG);
    }
#else
    const bf16t* A0 = Aseg[seg];
    #pragma unroll
    for (int i = 0; i < 2; ++i){
      int s = tid + i*256;
      int row = s >> 2, p = s & 3;
      const bf16t* srcA = A0 + (size_t)(mBase + row)*H1 + kk + p*8;
      const bf16t* srcB = Wb + (size_t)(nBase + row)*KBIG + ksBase + p*8;
      *(u32x4*)&As[buf][row][p*8] = *(const u32x4*)srcA;
      *(u32x4*)&Bs[buf][row][p*8] = *(const u32x4*)srcB;
      int ks2 = ksBase + 64;
      if (ks2 < KBIG){
        __builtin_prefetch(Aseg[ks2>>10] + (size_t)(mBase + row)*H1 + (ks2&1023) + p*8, 0, 1);
        __builtin_prefetch(Wb + (size_t)(nBase + row)*KBIG + ks2 + p*8, 0, 1);
      }
    }
#endif
  };
  auto stage_wait = [&](){
#if TDM_PATH
    if (wave == 0) __builtin_amdgcn_s_wait_tensorcnt(0);
#endif
  };

  v8f zero = {};
  v8f acc[4][2];
  #pragma unroll
  for (int a = 0; a < 4; ++a)
    #pragma unroll
    for (int b = 0; b < 2; ++b) acc[a][b] = zero;

  stage(0, 0);
  stage_wait();
  __syncthreads();

  const int mrow  = lane & 15;
  const int koffA = (lane < 16) ? 0 : 8;    // A 16x32: low lanes K0-7/16-23
  const int koffB = (lane < 16) ? 0 : 8;    // B 32x16: low lanes K0-7/16-23
  union Frag { v16bf v; u32x4 q[2]; };

  for (int kc = 0; kc < NK; ++kc){
    int buf = kc & 1;
    if (kc + 1 < NK) stage(kc + 1, buf ^ 1);

    Frag af[4], bfv[2];
    #pragma unroll
    for (int mf = 0; mf < 4; ++mf){
      int r = wm*64 + mf*16 + mrow;
      af[mf].q[0] = *(const u32x4*)&As[buf][r][koffA];
      af[mf].q[1] = *(const u32x4*)&As[buf][r][koffA + 16];
    }
    #pragma unroll
    for (int nf = 0; nf < 2; ++nf){
      int r = wn*32 + nf*16 + mrow;
      bfv[nf].q[0] = *(const u32x4*)&Bs[buf][r][koffB];
      bfv[nf].q[1] = *(const u32x4*)&Bs[buf][r][koffB + 16];
    }
    #pragma unroll
    for (int mf = 0; mf < 4; ++mf)
      #pragma unroll
      for (int nf = 0; nf < 2; ++nf)
        acc[mf][nf] = __builtin_amdgcn_wmma_f32_16x16x32_bf16(
            false, af[mf].v, false, bfv[nf].v, (short)0, acc[mf][nf], false, false);

    if (kc + 1 < NK) stage_wait();
    __syncthreads();
  }

  const int rHalf = (lane >> 4) * 8;
  #pragma unroll
  for (int mf = 0; mf < 4; ++mf){
    #pragma unroll
    for (int nf = 0; nf < 2; ++nf){
      int col  = nBase + wn*32 + nf*16 + mrow;
      int rowB = mBase + wm*64 + mf*16 + rHalf;
      float z = zterm[(size_t)blockIdx.x*NSMALL + col];   // batch = blockIdx.x
      #pragma unroll
      for (int r = 0; r < 8; ++r){
        int row = rowB + r;
        pre[(size_t)row*NBIG + col] = acc[mf][nf][r] + z;
      }
    }
  }
}

__global__ void k_gupdate(const float* __restrict__ pre, const float* __restrict__ smallout,
                          const float* __restrict__ ccur, const float* __restrict__ cg_old,
                          float* __restrict__ cg_new, float* __restrict__ g_new){
  int i = blockIdx.x*blockDim.x + threadIdx.x;
  if (i >= BB*H1) return;
  int b = i >> 10, k = i & 1023;
  const float* pp = pre  + (size_t)b*TT*NBIG + 6*H1 + k;
  const float* pc = ccur + (size_t)b*TT*H1 + k;
  float s = 0.f;
  for (int t = 0; t < TT; ++t)
    s += sigf(pp[(size_t)t*NBIG]) * pc[(size_t)t*H1];
  float gg = sigf(smallout[(size_t)b*NSMALL + NBIG + k]);
  float go = sigf(smallout[(size_t)b*NSMALL + NBIG + H1 + k]);
  float cgn = s + gg * cg_old[i];
  cg_new[i] = cgn;
  g_new[i]  = go * tanhf(cgn);
}

__global__ void k_hupdate(const float* __restrict__ pre, const float* __restrict__ chb,
                          const float* __restrict__ cha, const float* __restrict__ cg_old,
                          float* __restrict__ hcur, float* __restrict__ ccur){
  size_t i = (size_t)blockIdx.x*blockDim.x + threadIdx.x;
  if (i >= (size_t)BT*H1) return;
  int k = (int)(i & 1023);
  size_t row = i >> 10;
  int b = (int)(row >> 7);
  const float* p = pre + row*NBIG + k;
  float f  = sigf(p[0]);
  float l  = sigf(p[H1]);
  float r  = sigf(p[2*H1]);
  float q  = sigf(p[3*H1]);
  float ii = sigf(p[4*H1]);
  float o  = sigf(p[5*H1]);
  float cold = ccur[i];
  float cn = l*chb[i] + r*cha[i] + f*cold + q*cg_old[(size_t)b*H1 + k] + ii*cold;
  ccur[i] = cn;
  hcur[i] = o * tanhf(cn);
}

__global__ void k_final(const float* __restrict__ hcur, const float* __restrict__ ccur,
                        const float* __restrict__ g, float* __restrict__ out){
  size_t i = (size_t)blockIdx.x*blockDim.x + threadIdx.x;
  const size_t N = (size_t)BT*H1;
  if (i < N){ out[i] = hcur[i]; out[N + i] = ccur[i]; }
  if (i < (size_t)BB*H1) out[2*N + i] = g[i];
}

// ---------------- host ----------------

extern "C" void kernel_launch(void* const* d_in, const int* in_sizes, int n_in,
                              void* d_out, int out_size, void* d_ws, size_t ws_size,
                              hipStream_t stream) {
  const float* h   = (const float*)d_in[0];
  const float* c_h = (const float*)d_in[1];
  const float* U   = (const float*)d_in[2];
  const float* Wlr = (const float*)d_in[3];
  const float* W   = (const float*)d_in[4];
  const float* Z   = (const float*)d_in[5];
  const float* bb  = (const float*)d_in[6];
  const float* gZ  = (const float*)d_in[7];
  const float* gWp = (const float*)d_in[8];
  const float* gb  = (const float*)d_in[9];
  float* outp = (float*)d_out;

  char* ws = (char*)d_ws;
  size_t off = 0;
  auto carve = [&](size_t bytes)->char* {
    char* p = ws + off;
    off += (bytes + 255) & ~(size_t)255;
    return p;
  };

  bf16t* Wb       = (bf16t*)carve((size_t)NBIG*KBIG*sizeof(bf16t));
  bf16t* Wsm      = (bf16t*)carve((size_t)NSMALL*KSMALL*sizeof(bf16t));
  float* bsmall   = (float*)carve((size_t)NSMALL*sizeof(float));
  float* hcur     = (float*)carve((size_t)BT*H1*sizeof(float));
  float* ccur     = (float*)carve((size_t)BT*H1*sizeof(float));
  bf16t* hbf      = (bf16t*)carve((size_t)BT*H1*sizeof(bf16t));
  bf16t* hbbf     = (bf16t*)carve((size_t)BT*H1*sizeof(bf16t));
  bf16t* habf     = (bf16t*)carve((size_t)BT*H1*sizeof(bf16t));
  float* chb      = (float*)carve((size_t)BT*H1*sizeof(float));
  float* cha      = (float*)carve((size_t)BT*H1*sizeof(float));
  bf16t* gA       = (bf16t*)carve((size_t)BB*KSMALL*sizeof(bf16t));
  float* smallout = (float*)carve((size_t)BB*NSMALL*sizeof(float));
  float* pre      = (float*)carve((size_t)BT*NBIG*sizeof(float));
  float* gbuf0    = (float*)carve((size_t)BB*H1*sizeof(float));
  float* gbuf1    = (float*)carve((size_t)BB*H1*sizeof(float));
  float* cgbuf0   = (float*)carve((size_t)BB*H1*sizeof(float));
  float* cgbuf1   = (float*)carve((size_t)BB*H1*sizeof(float));
  float* gbuf[2]  = {gbuf0, gbuf1};
  float* cgbuf[2] = {cgbuf0, cgbuf1};
  (void)in_sizes; (void)n_in; (void)out_size; (void)ws_size;

  {
    size_t n = (size_t)NBIG*KBIG;
    k_prep_wb<<<(unsigned)((n + 255)/256), 256, 0, stream>>>(U, Wlr, W, gWp, Wb);
  }
  {
    size_t n = (size_t)NSMALL*KSMALL;
    k_prep_wsmall<<<(unsigned)((n + 255)/256), 256, 0, stream>>>(Z, gZ, gWp, Wsm);
  }
  k_prep_bsmall<<<(NSMALL + 255)/256, 256, 0, stream>>>(bb, gb, bsmall);
  {
    int n = BT*H1;
    k_copy2<<<(n + 255)/256, 256, 0, stream>>>(h, c_h, hcur, ccur, n);
  }
  k_means2<<<(BB*H1 + 255)/256, 256, 0, stream>>>(h, c_h, gbuf[0], cgbuf[0]);

  int cur = 0;
  for (int step = 0; step < REC; ++step){
    size_t nel = (size_t)BT*H1;
    k_winsum<<<(unsigned)((nel + 255)/256), 256, 0, stream>>>(hcur, ccur, hbf, hbbf, habf, chb, cha);
    k_build_gA<<<(BB*H1 + 255)/256, 256, 0, stream>>>(gbuf[cur], hcur, gA);
    k_gemm_small<<<(NSMALL/16)/8, 256, 0, stream>>>(gA, Wsm, bsmall, smallout);
    dim3 grid(BT/128, NBIG/128);
    k_gemm_main<<<grid, 256, 0, stream>>>(hbf, hbbf, habf, Wb, smallout, pre);
    k_gupdate<<<(BB*H1 + 255)/256, 256, 0, stream>>>(pre, smallout, ccur, cgbuf[cur],
                                                     cgbuf[cur^1], gbuf[cur^1]);
    k_hupdate<<<(unsigned)((nel + 255)/256), 256, 0, stream>>>(pre, chb, cha, cgbuf[cur], hcur, ccur);
    cur ^= 1;
  }
  {
    size_t n = (size_t)BT*H1;
    k_final<<<(unsigned)((n + 255)/256), 256, 0, stream>>>(hcur, ccur, gbuf[cur], outp);
  }
}
